// HyperGraphModel_62380105008306
// MI455X (gfx1250) — matrix operand order repaired and checked by reading, not compiled, not run on hardware
//
#include <hip/hip_runtime.h>
#include <hip/hip_bf16.h>

#ifndef __has_builtin
#define __has_builtin(x) 0
#endif

#define H_DIM 64
#define HD_DIM 32
#define N_PACKET 8192
#define N_ROUTER 16384
#define N_CHANNEL 65536
#define E_PASS 65536
#define N_CLASSES 11
#define NUM_MP 2

typedef __attribute__((ext_vector_type(16))) _Float16 v16h;
typedef __attribute__((ext_vector_type(8)))  float    v8f;

static __device__ __forceinline__ v8f wmma32(v16h a, v16h b, v8f c) {
    // D = A(16x32 f16) * B(32x16 f16) + C(16x16 f32)
    return __builtin_amdgcn_wmma_f32_16x16x32_f16(
        /*neg_a=*/false, a, /*neg_b=*/false, b,
        /*c_mod=*/(short)0, c, /*reuse_a=*/false, /*reuse_b=*/false);
}

// Within-chunk K offset for a 16-bit A-matrix fragment element (16x32 layout).
static __device__ __forceinline__ int koffA(int group, int e) {
    return group * 16 + e;
}

// ---- async global->LDS staging (CDNA5 path, guarded with fallback) ----
#if __has_builtin(__builtin_amdgcn_global_load_async_to_lds_b128)
#define HAVE_ASYNC_LDS 1
#else
#define HAVE_ASYNC_LDS 0
#endif

// Exact pointee type clang expects: int __attribute__((vector_size(16)))
typedef int v4i_vs __attribute__((vector_size(16)));
typedef __attribute__((address_space(1))) v4i_vs* as1_v4i_ptr;
typedef __attribute__((address_space(3))) v4i_vs* as3_v4i_ptr;

static __device__ __forceinline__ void stage16(void* lds_dst, const void* gsrc) {
#if HAVE_ASYNC_LDS
    __builtin_amdgcn_global_load_async_to_lds_b128(
        (as1_v4i_ptr)gsrc, (as3_v4i_ptr)lds_dst, 0, 0);
#else
    *(float4*)lds_dst = *(const float4*)gsrc;
#endif
}

static __device__ __forceinline__ void wait_async_done() {
#if HAVE_ASYNC_LDS
#if __has_builtin(__builtin_amdgcn_s_wait_asynccnt)
    __builtin_amdgcn_s_wait_asynccnt(0);
#else
    asm volatile("s_wait_asynccnt 0x0" ::: "memory");
#endif
#endif
}

// ---------------- utility kernels ----------------

__global__ void k_zero(float* __restrict__ p, int n) {
    int i = blockIdx.x * blockDim.x + threadIdx.x;
    if (i < n) p[i] = 0.0f;
}

// Pack a row-major f32 [K][N] matrix into WMMA-B fragment-major f16:
// dst[((kk*ntiles + nt)*32 + lane)*16 + e], B element: K = kk*32 + (lane/16)*16 + e,
// N = nt*16 + lane%16.
__global__ void k_pack_bfrag(const float* __restrict__ src, _Float16* __restrict__ dst,
                             int K, int N) {
    int idx = blockIdx.x * blockDim.x + threadIdx.x;
    int ntiles = N >> 4;
    int total = (K >> 5) * ntiles * 512;
    if (idx >= total) return;
    int e    = idx & 15;
    int lane = (idx >> 4) & 31;
    int nt   = (idx >> 9) % ntiles;
    int kk   = idx / (512 * ntiles);
    int kglob = kk * 32 + (lane >> 4) * 16 + e;
    int d     = nt * 16 + (lane & 15);
    dst[idx] = (_Float16)src[kglob * N + d];
}

// Pack w_mp[64][2048] + b_mp[2048] into the G-GEMM B matrix [4160][32] (frag-major).
// Logical row j = k*64 + h  ->  w_mp[k][h*32 + d]; rows 4096+h -> b_mp[h*32 + d].
__global__ void k_pack_wmp(const float* __restrict__ w_mp, const float* __restrict__ b_mp,
                           _Float16* __restrict__ dst) {
    int idx = blockIdx.x * blockDim.x + threadIdx.x;
    if (idx >= 130 * 2 * 512) return;
    int e    = idx & 15;
    int lane = (idx >> 4) & 31;
    int kk   = idx >> 10;
    int koff = (lane >> 4) * 16 + e;
    int d    = ((idx >> 9) & 1) * 16 + (lane & 15);
    float v;
    if (kk < 128) {
        int k = kk >> 1;
        int h = (kk & 1) * 32 + koff;
        v = w_mp[k * 2048 + h * 32 + d];
    } else {
        int h = (kk - 128) * 32 + koff;
        v = b_mp[h * 32 + d];
    }
    dst[idx] = (_Float16)v;
}

// ---------------- FeatureGen ----------------

__global__ __launch_bounds__(32) void k_packet(
    const float* __restrict__ freq, const float* __restrict__ flit,
    const float* __restrict__ w_freq, const float* __restrict__ b_freq,
    const float* __restrict__ w_flit, const float* __restrict__ b_flit,
    const _Float16* __restrict__ fragFP, const float* __restrict__ b_fp,
    float* __restrict__ packet_feat) {
    __shared__ _Float16 xs[16 * 128];
    int p0 = blockIdx.x * 16;
    int lane = threadIdx.x;
    for (int idx = lane; idx < 16 * 128; idx += 32) {
        int p = idx >> 7, col = idx & 127;
        float v;
        if (col < 64) {
            v = freq[p0 + p] * w_freq[col] + b_freq[col];
        } else {
            int c2 = col - 64;
            float a = b_flit[c2];
            #pragma unroll
            for (int k = 0; k < 32; k++) a += flit[(p0 + p) * 32 + k] * w_flit[k * 64 + c2];
            v = a;
        }
        xs[idx] = (_Float16)fmaxf(v, 0.0f);
    }
    __syncthreads();
    int group = lane >> 4, m = lane & 15;
    v8f zero = {};
    v8f acc[4];
    #pragma unroll
    for (int nt = 0; nt < 4; nt++) acc[nt] = zero;
    const v16h* B = (const v16h*)fragFP;
    #pragma unroll
    for (int kk = 0; kk < 4; kk++) {
        v16h a;
        #pragma unroll
        for (int e = 0; e < 16; e++) a[e] = xs[m * 128 + kk * 32 + koffA(group, e)];
        #pragma unroll
        for (int nt = 0; nt < 4; nt++)
            acc[nt] = wmma32(a, B[(kk * 4 + nt) * 32 + lane], acc[nt]);
    }
    #pragma unroll
    for (int nt = 0; nt < 4; nt++)
        #pragma unroll
        for (int r = 0; r < 8; r++) {
            int row = p0 + group * 8 + r, col = nt * 16 + m;
            packet_feat[row * 64 + col] = acc[nt][r] + b_fp[col];
        }
}

__global__ __launch_bounds__(32) void k_router(
    const float* __restrict__ op, const float* __restrict__ w_op,
    const float* __restrict__ b_op, const _Float16* __restrict__ fragFR,
    const float* __restrict__ b_fr, float* __restrict__ hidden) {
    __shared__ _Float16 xs[16 * 64];
    int r0 = blockIdx.x * 16;
    int lane = threadIdx.x;
    for (int idx = lane; idx < 16 * 64; idx += 32) {
        int p = idx >> 6, col = idx & 63;
        float a = b_op[col];
        #pragma unroll
        for (int k = 0; k < 4; k++) a += op[(r0 + p) * 4 + k] * w_op[k * 64 + col];
        xs[idx] = (_Float16)fmaxf(a, 0.0f);
    }
    __syncthreads();
    int group = lane >> 4, m = lane & 15;
    v8f zero = {};
    v8f acc[4];
    #pragma unroll
    for (int nt = 0; nt < 4; nt++) acc[nt] = zero;
    const v16h* B = (const v16h*)fragFR;
    #pragma unroll
    for (int kk = 0; kk < 2; kk++) {
        v16h a;
        #pragma unroll
        for (int e = 0; e < 16; e++) a[e] = xs[m * 64 + kk * 32 + koffA(group, e)];
        #pragma unroll
        for (int nt = 0; nt < 4; nt++)
            acc[nt] = wmma32(a, B[(kk * 4 + nt) * 32 + lane], acc[nt]);
    }
    #pragma unroll
    for (int nt = 0; nt < 4; nt++)
        #pragma unroll
        for (int r = 0; r < 8; r++) {
            int row = r0 + group * 8 + r, col = nt * 16 + m;
            hidden[row * 64 + col] = acc[nt][r] + b_fr[col];
        }
}

// A[c] = sum over pass edges into c of packet_feat[src]; deg[c] = edge count.
__global__ void k_scatterA(const int* __restrict__ pass_src, const int* __restrict__ pass_dst,
                           const float* __restrict__ packet_feat,
                           float* __restrict__ A, float* __restrict__ deg) {
    int idx = blockIdx.x * blockDim.x + threadIdx.x;
    if (idx >= E_PASS * 64) return;
    int e = idx >> 6, f = idx & 63;
    int s = pass_src[e], d = pass_dst[e];
    atomicAdd(&A[d * 64 + f], packet_feat[s * 64 + f]);
    if (f == 0) atomicAdd(&deg[d], 1.0f);
}

// ---------------- fused message-passing einsum ----------------
// Block = 64 channels, 256 threads = 8 waves: wave w -> channel-tile w>>1,
// direction w&1. The whole 266 KB fragment-packed w_mp (B operand) plus the
// 16 KB A-block are staged ONCE per block into dynamic LDS (320 KB/WGP budget)
// with async global->LDS copies; the K-loop then runs entirely on ds_load_b128
// + v_wmma. Results scattered to m_in/m_out halves of mbuf with f32 atomics.
#define WFRAG_BYTES (130 * 2 * 512 * 2)          // 266240
#define EIN_AS_OFF  WFRAG_BYTES                  // 64*64 f32 = 16384
#define EIN_DEG_OFF (WFRAG_BYTES + 16384)        // 64 f32
#define EIN_SMEM    (WFRAG_BYTES + 16384 + 256)  // 282880 <= 320K

__global__ __launch_bounds__(256) void k_einsum(
    const float* __restrict__ hidden, const float* __restrict__ A,
    const float* __restrict__ deg, const int* __restrict__ out_src,
    const int* __restrict__ in_dst, const _Float16* __restrict__ fragMP,
    float* __restrict__ mbuf) {
    extern __shared__ char smem[];
    _Float16* Bs  = (_Float16*)smem;
    float*    As  = (float*)(smem + EIN_AS_OFF);
    float*    dgs = (float*)(smem + EIN_DEG_OFF);

    int c0 = blockIdx.x * 64;
    int tid = threadIdx.x;

    // stage B fragments: 266240 B = 16640 x 16B
    for (int i = tid; i < WFRAG_BYTES / 16; i += 256)
        stage16(Bs + i * 8, fragMP + i * 8);
    // stage A block: 64 ch x 64 f32 = 1024 x 16B
    for (int i = tid; i < 1024; i += 256)
        stage16(As + i * 4, A + (size_t)c0 * 64 + i * 4);
    // stage degrees: 64 f32 = 16 x 16B
    if (tid < 16)
        stage16(dgs + tid * 4, deg + c0 + tid * 4);
    wait_async_done();
    __syncthreads();

    int wave = tid >> 5;
    int dir  = wave & 1;          // 0 = in, 1 = out (uniform per wave)
    int ct   = wave >> 1;         // channel tile 0..3
    int lane = tid & 31;
    int group = lane >> 4, m = lane & 15;
    int lc = ct * 16 + m;         // local channel for A/deg
    int c  = c0 + lc;
    int src = (dir == 0) ? out_src[c] : in_dst[c];   // rfeat gather row
    const float* hrow = hidden + src * 64;

    v16h rf0, rf1;                 // rfeat fragments for even/odd K-chunks
    #pragma unroll
    for (int e = 0; e < 16; e++) {
        rf0[e] = (_Float16)hrow[0 * 32 + koffA(group, e)];
        rf1[e] = (_Float16)hrow[1 * 32 + koffA(group, e)];
    }

    v8f acc0 = {}, acc1 = {};
    const v16h* Bv = (const v16h*)Bs;
    for (int kk = 0; kk < 130; kk++) {
        float s = (kk < 128) ? As[lc * 64 + (kk >> 1)] : dgs[lc];
        _Float16 hs = (_Float16)s;
        v16h rq = (kk & 1) ? rf1 : rf0;
        v16h a;
        #pragma unroll
        for (int e = 0; e < 16; e++) a[e] = hs * rq[e];   // v_pk_mul_f16
        v16h b0 = Bv[(kk * 2 + 0) * 32 + lane];
        v16h b1 = Bv[(kk * 2 + 1) * 32 + lane];
        acc0 = wmma32(a, b0, acc0);
        acc1 = wmma32(a, b1, acc1);
    }

    #pragma unroll
    for (int r = 0; r < 8; r++) {
        int cm = c0 + ct * 16 + group * 8 + r;
        int rr = (dir == 0) ? in_dst[cm] : out_src[cm];   // router scatter row
        float* dst = mbuf + rr * 64 + dir * 32;
        atomicAdd(&dst[0 * 16 + m], acc0[r]);
        atomicAdd(&dst[1 * 16 + m], acc1[r]);
    }
}

__global__ void k_update(float* __restrict__ hidden, const float* __restrict__ mbuf, int n) {
    int i = blockIdx.x * blockDim.x + threadIdx.x;
    if (i < n) hidden[i] = fmaxf(hidden[i] + mbuf[i], 0.0f);
}

// ---------------- readout + head ----------------

__global__ __launch_bounds__(64) void k_colsum(const float* __restrict__ hidden,
                                               float* __restrict__ embed) {
    int col = threadIdx.x;
    float s = 0.0f;
    for (int r = blockIdx.x; r < N_ROUTER; r += gridDim.x) s += hidden[r * 64 + col];
    atomicAdd(&embed[col], s);
}

__global__ __launch_bounds__(64) void k_head(
    const float* __restrict__ embed,
    const float* __restrict__ w_h1, const float* __restrict__ b_h1,
    const float* __restrict__ w_h2, const float* __restrict__ b_h2,
    const float* __restrict__ w_out, const float* __restrict__ b_out,
    float* __restrict__ out) {
    __shared__ float s0[64], s1[64], s2[64];
    int t = threadIdx.x;
    s0[t] = embed[t];
    __syncthreads();
    float a = b_h1[t];
    for (int k = 0; k < 64; k++) a += s0[k] * w_h1[k * 64 + t];
    s1[t] = fmaxf(a, 0.0f);
    __syncthreads();
    a = b_h2[t];
    for (int k = 0; k < 64; k++) a += s1[k] * w_h2[k * 64 + t];
    s2[t] = fmaxf(a, 0.0f);
    __syncthreads();
    if (t < N_CLASSES) {
        a = b_out[t];
        for (int k = 0; k < 64; k++) a += s2[k] * w_out[k * N_CLASSES + t];
        out[t] = a;
    }
}

// ---------------- host launch ----------------

extern "C" void kernel_launch(void* const* d_in, const int* in_sizes, int n_in,
                              void* d_out, int out_size, void* d_ws, size_t ws_size,
                              hipStream_t stream) {
    (void)in_sizes; (void)n_in; (void)out_size; (void)ws_size;
    const float* freq     = (const float*)d_in[0];
    const float* flit     = (const float*)d_in[1];
    const float* op_type  = (const float*)d_in[2];
    const int*   out_src  = (const int*)d_in[4];
    const int*   in_dst   = (const int*)d_in[7];
    const int*   pass_src = (const int*)d_in[8];
    const int*   pass_dst = (const int*)d_in[9];
    const float* w_freq = (const float*)d_in[10];
    const float* b_freq = (const float*)d_in[11];
    const float* w_flit = (const float*)d_in[12];
    const float* b_flit = (const float*)d_in[13];
    const float* w_op   = (const float*)d_in[14];
    const float* b_op   = (const float*)d_in[15];
    const float* w_fp   = (const float*)d_in[18];
    const float* b_fp   = (const float*)d_in[19];
    const float* w_fr   = (const float*)d_in[20];
    const float* b_fr   = (const float*)d_in[21];
    const float* w_mp   = (const float*)d_in[24];
    const float* b_mp   = (const float*)d_in[25];
    const float* w_h1   = (const float*)d_in[26];
    const float* b_h1   = (const float*)d_in[27];
    const float* w_h2   = (const float*)d_in[28];
    const float* b_h2   = (const float*)d_in[29];
    const float* w_out  = (const float*)d_in[30];
    const float* b_out  = (const float*)d_in[31];

    char* ws = (char*)d_ws;
    auto alloc = [&](size_t bytes) {
        char* p = ws;
        ws += (bytes + 255) & ~(size_t)255;
        return p;
    };
    float*     packet_feat = (float*)alloc((size_t)N_PACKET * 64 * 4);
    float*     hidden      = (float*)alloc((size_t)N_ROUTER * 64 * 4);
    float*     Aagg        = (float*)alloc((size_t)N_CHANNEL * 64 * 4);
    float*     deg         = (float*)alloc((size_t)N_CHANNEL * 4);
    float*     mbuf        = (float*)alloc((size_t)N_ROUTER * 64 * 4);
    _Float16*  fragMP      = (_Float16*)alloc((size_t)WFRAG_BYTES);
    _Float16*  fragFP      = (_Float16*)alloc((size_t)4 * 4 * 512 * 2);
    _Float16*  fragFR      = (_Float16*)alloc((size_t)2 * 4 * 512 * 2);
    float*     embed       = (float*)alloc(64 * 4);

    // zero scatter targets
    k_zero<<<(N_CHANNEL * 64 + 255) / 256, 256, 0, stream>>>(Aagg, N_CHANNEL * 64);
    k_zero<<<(N_CHANNEL + 255) / 256, 256, 0, stream>>>(deg, N_CHANNEL);

    // one-time weight repacks into WMMA-B fragment layout
    k_pack_wmp<<<(130 * 2 * 512 + 255) / 256, 256, 0, stream>>>(w_mp, b_mp, fragMP);
    k_pack_bfrag<<<(4 * 4 * 512 + 255) / 256, 256, 0, stream>>>(w_fp, fragFP, 128, 64);
    k_pack_bfrag<<<(2 * 4 * 512 + 255) / 256, 256, 0, stream>>>(w_fr, fragFR, 64, 64);

    // FeatureGen
    k_packet<<<N_PACKET / 16, 32, 0, stream>>>(freq, flit, w_freq, b_freq, w_flit, b_flit,
                                               fragFP, b_fp, packet_feat);
    k_router<<<N_ROUTER / 16, 32, 0, stream>>>(op_type, w_op, b_op, fragFR, b_fr, hidden);

    // channel-level packet aggregate (iteration-invariant)
    k_scatterA<<<(E_PASS * 64 + 255) / 256, 256, 0, stream>>>(pass_src, pass_dst,
                                                              packet_feat, Aagg, deg);

    // message passing
    for (int it = 0; it < NUM_MP; it++) {
        k_zero<<<(N_ROUTER * 64 + 255) / 256, 256, 0, stream>>>(mbuf, N_ROUTER * 64);
        k_einsum<<<N_CHANNEL / 64, 256, EIN_SMEM, stream>>>(hidden, Aagg, deg, out_src,
                                                            in_dst, fragMP, mbuf);
        k_update<<<(N_ROUTER * 64 + 255) / 256, 256, 0, stream>>>(hidden, mbuf,
                                                                  N_ROUTER * 64);
    }

    // readout + prediction head
    k_zero<<<1, 64, 0, stream>>>(embed, 64);
    k_colsum<<<64, 64, 0, stream>>>(hidden, embed);
    k_head<<<1, 64, 0, stream>>>(embed, w_h1, b_h1, w_h2, b_h2, w_out, b_out,
                                 (float*)d_out);
}
